// OccupancyFlow_Field_v2_56745107915346
// MI455X (gfx1250) — matrix operand order repaired and checked by reading, not compiled, not run on hardware
//
#include <hip/hip_runtime.h>
#include <hip/hip_bf16.h>

// ---------------------------------------------------------------------------
// PointPillars-style pipeline for MI455X (gfx1250, wave32, WMMA).
//   ws layout: [ bf16 canvas 8*256*256*128 (128 MB) | bf16 packed conv weights
//               36 ksteps * 2 ntiles * 32 lanes * 16 el (72 KB) ]
// ---------------------------------------------------------------------------

typedef __attribute__((ext_vector_type(16))) __bf16 v16bf;
typedef __attribute__((ext_vector_type(8)))  float  v8f;

#define NXc 256
#define NYc 256
#define Bc  8
#define CIN 128   // concatenated canvas channels
#define COUT 32
#define KSTEPS 36 // 9 taps * (128/32)

__device__ __forceinline__ unsigned short f2bf(float f) {
    union { float f; unsigned u; } c; c.f = f;
    unsigned u = c.u;
    u += 0x7fffu + ((u >> 16) & 1u);   // round-to-nearest-even
    return (unsigned short)(u >> 16);
}

// ---------------------------------------------------------------------------
// 0) zero the bf16 canvas (grid-stride, 16B stores)
// ---------------------------------------------------------------------------
__global__ void zero_canvas_kernel(uint4* __restrict__ p, size_t n16) {
    size_t tid = (size_t)blockIdx.x * blockDim.x + threadIdx.x;
    size_t stride = (size_t)gridDim.x * blockDim.x;
    uint4 z = {0u, 0u, 0u, 0u};
    for (size_t i = tid; i < n16; i += stride) p[i] = z;
}

// ---------------------------------------------------------------------------
// 1) pack conv_w (3,3,128,32) f32 -> bf16 WMMA B-fragments.
//    thread t = ((s*2)+nt)*32 + lane ; element j -> K = (lane>>4)*8 + (j&7) + (j>>3)*16
// ---------------------------------------------------------------------------
__global__ void pack_weights_kernel(const float* __restrict__ conv_w,
                                    unsigned short* __restrict__ fragW) {
    int t = blockIdx.x * blockDim.x + threadIdx.x;
    if (t >= KSTEPS * 2 * 32) return;
    int lane = t & 31;
    int nt   = (t >> 5) & 1;
    int s    = t >> 6;
    int n    = nt * 16 + (lane & 15);
    unsigned short* dst = fragW + (size_t)t * 16;
    #pragma unroll
    for (int j = 0; j < 16; ++j) {
        int klocal = ((lane >> 4) << 3) + (j & 7) + ((j >> 3) << 4);   // 0..31
        int k  = s * 32 + klocal;              // 0..1151
        int tap = k >> 7;                      // ky*3+kx
        int ci  = k & 127;
        dst[j] = f2bf(conv_w[((size_t)tap * CIN + ci) * COUT + n]);
    }
}

// ---------------------------------------------------------------------------
// 2) PillarVFE + scatter. One wave32 per pillar (P<=32), lane = point index.
//    Weight matrix (F x 64) staged in LDS -> ds_load traffic.
// ---------------------------------------------------------------------------
__global__ void vfe_scatter_kernel(const float* __restrict__ voxels,
                                   const int*   __restrict__ coors,
                                   const int*   __restrict__ npoints,
                                   const float* __restrict__ W,
                                   const float* __restrict__ bias,
                                   unsigned short* __restrict__ canvas,
                                   int M, int P, int C, int ch_off) {
    __shared__ float sW[10 * 64];
    __shared__ float sB[64];
    const int F = C + 5;
    for (int i = threadIdx.x; i < F * 64; i += blockDim.x) sW[i] = W[i];
    if (threadIdx.x < 64) sB[threadIdx.x] = bias[threadIdx.x];
    __syncthreads();

    const int wave = threadIdx.x >> 5;
    const int lane = threadIdx.x & 31;
    const int pillar = blockIdx.x * 8 + wave;
    if (pillar >= M) return;

    const int np = npoints[pillar];
    const float npf = (float)(np > 0 ? np : 1);
    const int p = lane;
    const bool valid = (p < P) && (p < np);
    const float mask = valid ? 1.0f : 0.0f;

    float v[5];
    #pragma unroll
    for (int k = 0; k < 5; ++k) v[k] = 0.0f;
    if (p < P) {
        const float* src = voxels + ((size_t)pillar * P + p) * C;
        for (int k = 0; k < C; ++k) v[k] = src[k];
    }

    // masked mean of xyz over points (butterfly sum across wave32)
    float sx = v[0] * mask, sy = v[1] * mask, sz = v[2] * mask;
    #pragma unroll
    for (int off = 16; off > 0; off >>= 1) {
        sx += __shfl_xor(sx, off, 32);
        sy += __shfl_xor(sy, off, 32);
        sz += __shfl_xor(sz, off, 32);
    }
    const float mx = sx / npf, my = sy / npf, mz = sz / npf;

    const int bi  = coors[pillar * 3 + 0];
    const int cyi = coors[pillar * 3 + 1];
    const int cxi = coors[pillar * 3 + 2];

    const float VX = 1.0f / 3.2f, VY = 1.0f / 3.2f;
    const float X_OFF = VX * 0.5f - 40.0f, Y_OFF = VY * 0.5f - 20.0f;

    float f[10];
    for (int k = 0; k < C; ++k) f[k] = v[k] * mask;
    f[C + 0] = (v[0] - mx) * mask;
    f[C + 1] = (v[1] - my) * mask;
    f[C + 2] = (v[2] - mz) * mask;
    f[C + 3] = (v[0] - ((float)cxi * VX + X_OFF)) * mask;
    f[C + 4] = (v[1] - ((float)cyi * VY + Y_OFF)) * mask;

    const size_t base = (((size_t)bi * NYc + cyi) * NXc + cxi) * CIN + ch_off;

    for (int co = 0; co < 64; ++co) {
        float acc = sB[co];
        for (int k = 0; k < F; ++k) acc = fmaf(f[k], sW[k * 64 + co], acc);
        acc = fmaxf(acc, 0.0f) * mask;           // relu then point mask
        #pragma unroll
        for (int off = 16; off > 0; off >>= 1)   // max over points
            acc = fmaxf(acc, __shfl_xor(acc, off, 32));
        if (lane == 0) canvas[base + co] = f2bf(acc);
    }
}

// ---------------------------------------------------------------------------
// 3) 3x3 conv (128 -> 32) as implicit GEMM with v_wmma_f32_16x16x32_bf16.
//    Wave owns 16 consecutive x-pixels and both N-tiles; 72 WMMAs per wave.
//    Software-pipelined: step s+1's fragment loads issue before step s's
//    WMMAs so VMEM overlaps the matrix pipe instead of s_wait'ing each step.
// ---------------------------------------------------------------------------
struct Frag { union { v16bf v; uint4 q[2]; }; };

__global__ void __launch_bounds__(256)
conv3x3_wmma_kernel(const unsigned short* __restrict__ canvas,
                    const unsigned short* __restrict__ fragW,
                    const float* __restrict__ conv_b,
                    float* __restrict__ out) {
    const int wave = threadIdx.x >> 5;
    const int lane = threadIdx.x & 31;
    const int m    = lane & 15;     // pixel within wave tile / column n
    const int hi   = lane >> 4;     // K-group selector (A) / row-half (D)

    // pixel block: 128 consecutive pixels per block, 16 per wave (same b,y)
    const size_t pix0block = (size_t)blockIdx.x * 128;
    const int bb  = (int)(pix0block >> 16);
    const int yy0 = (int)((pix0block >> 8) & 255);
    const int xx0 = (int)(pix0block & 255) + wave * 16;

    v8f acc0 = {0.f, 0.f, 0.f, 0.f, 0.f, 0.f, 0.f, 0.f};
    v8f acc1 = {0.f, 0.f, 0.f, 0.f, 0.f, 0.f, 0.f, 0.f};

    auto loadA = [&](int s, Frag& a) {
        const int tap = s >> 2;           // ky*3+kx
        const int ky = tap / 3, kx = tap - 3 * ky;
        const int c0 = (s & 3) * 32;
        const int px = xx0 + m + kx - 1;
        const int py = yy0 + ky - 1;
        a.q[0] = make_uint4(0u, 0u, 0u, 0u);
        a.q[1] = make_uint4(0u, 0u, 0u, 0u);
        if (px >= 0 && px < NXc && py >= 0 && py < NYc) {
            const uint4* src = (const uint4*)(canvas +
                ((((size_t)bb * NYc + py) * NXc + px) * CIN + c0 + hi * 8));
            a.q[0] = src[0];   // K = c0 + hi*8 + {0..7}
            a.q[1] = src[2];   // K = c0 + hi*8 + 16 + {0..7}
            if (c0 == 96 && py + 1 < NYc)   // prefetch the row two below (py+2)
                __builtin_prefetch((const void*)(src + 2 * NXc * CIN / 8), 0, 0);
        }
    };
    auto loadB = [&](int s, Frag& b0, Frag& b1) {
        const uint4* bp0 = (const uint4*)fragW + ((size_t)(s * 2 + 0) * 32 + lane) * 2;
        const uint4* bp1 = (const uint4*)fragW + ((size_t)(s * 2 + 1) * 32 + lane) * 2;
        b0.q[0] = bp0[0]; b0.q[1] = bp0[1];
        b1.q[0] = bp1[0]; b1.q[1] = bp1[1];
    };

    Frag a, b0, b1, an, b0n, b1n;
    loadA(0, a);
    loadB(0, b0, b1);

    #pragma unroll 4
    for (int s = 0; s < KSTEPS; ++s) {
        if (s + 1 < KSTEPS) {             // issue next step's loads first
            loadA(s + 1, an);
            loadB(s + 1, b0n, b1n);
        }
        acc0 = __builtin_amdgcn_wmma_f32_16x16x32_bf16(
            false, a.v, false, b0.v, (short)0, acc0, false, false);
        acc1 = __builtin_amdgcn_wmma_f32_16x16x32_bf16(
            false, a.v, false, b1.v, (short)0, acc1, false, false);
        a = an; b0 = b0n; b1 = b1n;       // rotate double buffers
    }

    // ---- store D (+ conv bias). lane holds col n, rows hi*8 + r.
    const int n = m;
    const float cb0 = conv_b[n];
    const float cb1 = conv_b[16 + n];
    const size_t pixbase = pix0block + (size_t)wave * 16;
    #pragma unroll
    for (int r = 0; r < 8; ++r) {
        const size_t pix = pixbase + hi * 8 + r;
        out[pix * COUT + n]      = acc0[r] + cb0;
        out[pix * COUT + 16 + n] = acc1[r] + cb1;
    }
}

// ---------------------------------------------------------------------------
extern "C" void kernel_launch(void* const* d_in, const int* in_sizes, int n_in,
                              void* d_out, int out_size, void* d_ws, size_t ws_size,
                              hipStream_t stream) {
    const float* voxels_a = (const float*)d_in[0];
    const int*   coors_a  = (const int*)  d_in[1];
    const int*   np_a     = (const int*)  d_in[2];
    const float* voxels_r = (const float*)d_in[3];
    const int*   coors_r  = (const int*)  d_in[4];
    const int*   np_r     = (const int*)  d_in[5];
    const float* w_a      = (const float*)d_in[6];
    const float* b_a      = (const float*)d_in[7];
    const float* w_r      = (const float*)d_in[8];
    const float* b_r      = (const float*)d_in[9];
    const float* conv_w   = (const float*)d_in[10];
    const float* conv_b   = (const float*)d_in[11];

    const int Ma = in_sizes[2];
    const int Mr = in_sizes[5];

    unsigned short* canvas = (unsigned short*)d_ws;
    const size_t canvasElems = (size_t)Bc * NYc * NXc * CIN;   // 67,108,864 bf16
    unsigned short* fragW = canvas + canvasElems;

    zero_canvas_kernel<<<4096, 256, 0, stream>>>((uint4*)canvas, canvasElems / 8);
    pack_weights_kernel<<<(KSTEPS * 2 * 32 + 255) / 256, 256, 0, stream>>>(conv_w, fragW);

    vfe_scatter_kernel<<<(Ma + 7) / 8, 256, 0, stream>>>(
        voxels_a, coors_a, np_a, w_a, b_a, canvas, Ma, 20, 5, 0);
    vfe_scatter_kernel<<<(Mr + 7) / 8, 256, 0, stream>>>(
        voxels_r, coors_r, np_r, w_r, b_r, canvas, Mr, 32, 3, 64);

    conv3x3_wmma_kernel<<<(Bc * NYc * NXc) / 128, 256, 0, stream>>>(
        canvas, fragW, conv_b, (float*)d_out);
}
